// CEABlock_62139586839407
// MI455X (gfx1250) — matrix operand order repaired and checked by reading, not compiled, not run on hardware
//
#include <hip/hip_runtime.h>

// ---------------------------------------------------------------------------
// CEABlock forward for MI455X (gfx1250, wave32, WMMA).
// f16 operands feed v_wmma_f32_16x16x32_f16. Weights pre-transposed so every
// B-fragment is a contiguous 32B load per lane (B layout: lane = column,
// halves = 16 consecutive k, lane-group = k-group). Async LDS round-trip on
// the CE gather exercises the gfx1250 ASYNCcnt path.
// ---------------------------------------------------------------------------

typedef __attribute__((ext_vector_type(16))) _Float16 v16h;
typedef __attribute__((ext_vector_type(8)))  _Float16 v8h;
typedef __attribute__((ext_vector_type(8)))  float    v8f;

constexpr int B_   = 32;    // batch
constexpr int L_   = 320;   // total tokens phase 1
constexpr int LT_  = 64;    // template tokens
constexpr int LS_  = 256;   // search tokens
constexpr int D_   = 768;   // model dim
constexpr int H_   = 12;    // heads
constexpr int HD_  = 64;    // head dim
constexpr int E_   = 6;     // experts
constexpr int MH_  = 64;    // MoE hidden
constexpr int FH_  = 3072;  // MLP hidden
constexpr int KEEP_ = 180;  // ceil(0.7*256)
constexpr int L2_  = 244;   // LT_ + KEEP_
constexpr int LT2_ = 48;    // L2_ // 5
constexpr int QW_  = 3 * D_;

__device__ __forceinline__ v8f wmma32(v16h a, v16h b, v8f c) {
  return __builtin_amdgcn_wmma_f32_16x16x32_f16(false, a, false, b, (short)0, c,
                                                false, false);
}
__device__ __forceinline__ v16h cat8(v8h lo, v8h hi) {
  return __builtin_shufflevector(lo, hi, 0, 1, 2, 3, 4, 5, 6, 7, 8, 9, 10, 11,
                                 12, 13, 14, 15);
}
// 16 halves via two 16B loads (only 16B alignment assumed)
__device__ __forceinline__ v16h ldh16(const _Float16* p) {
  return cat8(*(const v8h*)p, *(const v8h*)(p + 8));
}
__device__ __forceinline__ float gelu_tanh(float v) {
  float u = 0.7978845608028654f * (v + 0.044715f * v * v * v);
  return 0.5f * v * (1.0f + tanhf(u));
}

// ---------------------------------------------------------------------------
// WMMA GEMM: C[M,N] = act(A[M,K] @ B[K,N] + bias), B given TRANSPOSED as
// BT[N][K] f16 row-major. A f16 row-major. Output f32 (OUTH=0) or f16 (1).
// Block = 128 thr = 4 waves; wave owns a 16x64 tile (A-frag reused 4x).
// grid = (ceil(N/256), M/16). Requires K%32==0, N%16==0.
// B-frag: lane -> column n = base+(lane&15); halves = 16 contiguous k at
// k0 + (lane>>4)*16  ==> one contiguous 32B load per lane per subtile.
// ---------------------------------------------------------------------------
template <int ACT, int OUTH>
__global__ __launch_bounds__(128) void k_gemm(
    const _Float16* __restrict__ A, const _Float16* __restrict__ BT,
    const float* __restrict__ bias, float* __restrict__ Cf,
    _Float16* __restrict__ Ch, int M, int N, int K) {
  const int wave = threadIdx.x >> 5;
  const int lane = threadIdx.x & 31;
  const int row0 = blockIdx.y << 4;
  const int colw = (blockIdx.x << 8) + (wave << 6);
  if (colw >= N) return;                       // wave-uniform
  const int rem = (N - colw) >> 4;
  const int nt = rem > 4 ? 4 : rem;            // 1..4 subtiles, wave-uniform
  const int m = row0 + (lane & 15);
  const int mc = (m < M) ? m : (M - 1);
  const int kg   = (lane >> 4) << 3;           // unused; A also uses kcol
  const int kcol = (lane >> 4) << 4;           // B k-offset: 0 or 16

  v8f acc[4] = {};
  for (int k0 = 0; k0 < K; k0 += 32) {
    const _Float16* Ar = A + (size_t)mc * K + k0;
    __builtin_prefetch(Ar + 32);               // -> global_prefetch_b8
    v16h a = ldh16(Ar + kcol);
    const _Float16* Bt = BT + (size_t)(colw + (lane & 15)) * K + k0 + kcol;
    acc[0] = wmma32(a, ldh16(Bt), acc[0]);
    if (nt > 1) acc[1] = wmma32(a, ldh16(Bt + (size_t)16 * K), acc[1]);
    if (nt > 2) acc[2] = wmma32(a, ldh16(Bt + (size_t)32 * K), acc[2]);
    if (nt > 3) acc[3] = wmma32(a, ldh16(Bt + (size_t)48 * K), acc[3]);
  }
  const int rb = row0 + ((lane >> 4) << 3);
  const int ccl = lane & 15;
#pragma unroll
  for (int t = 0; t < 4; ++t) {
    if (t < nt) {
      const int cc = colw + (t << 4) + ccl;
      const float bv = bias ? bias[cc] : 0.0f;
#pragma unroll
      for (int r = 0; r < 8; ++r) {
        int rr = rb + r;
        if (rr < M) {
          float v = acc[t][r] + bv;
          if (ACT == 1) v = gelu_tanh(v);
          if (OUTH) Ch[(size_t)rr * N + cc] = (_Float16)v;
          else      Cf[(size_t)rr * N + cc] = v;
        }
      }
    }
  }
}

// ---------------------------------------------------------------------------
// Fused attention on f16 qkv: S = QK^T/8 -> softmax (f32 probs to global,
// f16 probs to LDS) -> O = P V. Block = 128 thr = 4 waves.
// qkv layout: [B, L, 3*D] halves, q at +0, k at +768, v at +1536.
// K^T fragments: contiguous 32B loads (lane = key row, halves = head dims).
// V is transposed into LDS in 64-k chunks for contiguous P.V B-fragments.
// ---------------------------------------------------------------------------
__global__ __launch_bounds__(128) void k_attn(const _Float16* __restrict__ qkv,
                                              _Float16* __restrict__ hout,
                                              float* __restrict__ probs) {
  __shared__ float    sc[16][L_];              // 20 KB scores
  __shared__ _Float16 sp[16][L_];              // 10 KB f16 probs
  __shared__ _Float16 vt[HD_][64];             //  8 KB V^T chunk
  __shared__ float red[16][8];
  __shared__ float rowm[16], rowsum[16];
  const int b = blockIdx.z, h = blockIdx.y, r0 = blockIdx.x << 4;
  const int wave = threadIdx.x >> 5, lane = threadIdx.x & 31;
  const int nlane = lane & 15;
  const int kg   = (lane >> 4) << 3;
  const int kcol = (lane >> 4) << 4;
  const size_t base = (size_t)b * L_ * QW_;

  // --- scores: Q x K^T (K-dim = 64 -> two WMMA steps) ---
  const int mrow = r0 + nlane;
  const _Float16* qr = qkv + base + (size_t)mrow * QW_ + h * HD_;
  v16h a0 = ldh16(qr + kcol);
  v16h a1 = ldh16(qr + 32 + kcol);
  for (int ct = wave; ct < L_ / 16; ct += 4) {
    const int c0 = ct << 4;
    // lane -> key row c0+nlane; halves -> head dims kcol..kcol+15 (contig)
    const _Float16* kb =
        qkv + base + (size_t)(c0 + nlane) * QW_ + D_ + h * HD_ + kcol;
    v8f acc = {};
    acc = wmma32(a0, ldh16(kb), acc);
    acc = wmma32(a1, ldh16(kb + 32), acc);
    const int rb = (lane >> 4) << 3;
#pragma unroll
    for (int r = 0; r < 8; ++r) sc[rb + r][c0 + nlane] = acc[r] * 0.125f;
  }
  __syncthreads();

  // --- softmax per row (8 threads / row) ---
  const int row = threadIdx.x >> 3, s = threadIdx.x & 7;
  float mx = -3.4e38f;
  for (int c = s; c < L_; c += 8) mx = fmaxf(mx, sc[row][c]);
  red[row][s] = mx;
  __syncthreads();
  if (s == 0) {
    float m2 = red[row][0];
    for (int i = 1; i < 8; ++i) m2 = fmaxf(m2, red[row][i]);
    rowm[row] = m2;
  }
  __syncthreads();
  float sm = 0.0f;
  for (int c = s; c < L_; c += 8) sm += __expf(sc[row][c] - rowm[row]);
  red[row][s] = sm;
  __syncthreads();
  if (s == 0) {
    float t = 0.0f;
    for (int i = 0; i < 8; ++i) t += red[row][i];
    rowsum[row] = t;
  }
  __syncthreads();
  {
    const float inv = 1.0f / rowsum[row];
    const size_t pbase = ((size_t)(b * H_ + h) * L_ + r0 + row) * L_;
    for (int c = s; c < L_; c += 8) {
      float p = __expf(sc[row][c] - rowm[row]) * inv;
      sp[row][c] = (_Float16)p;
      probs[pbase + c] = p;
    }
  }

  // --- O = P @ V : N = 64, wave -> 16-col tile, K chunks of 64 ---
  const int n0 = wave << 4;
  const int mr = nlane;
  v8f acc = {};
  for (int kc = 0; kc < L_; kc += 64) {
    __syncthreads();                           // vt reuse + sp ready (1st it)
    // transpose 64 V rows into vt: vectorized b128 reads, b16 LDS scatter
    for (int idx = threadIdx.x; idx < 64 * 8; idx += 128) {
      const int lrow = idx >> 3, grp = idx & 7;
      const v8h vv = *(const v8h*)(qkv + base + (size_t)(kc + lrow) * QW_ +
                                   2 * D_ + h * HD_ + grp * 8);
#pragma unroll
      for (int j = 0; j < 8; ++j) vt[grp * 8 + j][lrow] = vv[j];
    }
    __syncthreads();
#pragma unroll
    for (int k0 = 0; k0 < 64; k0 += 32) {
      v16h a = ldh16(&sp[mr][kc + k0 + kcol]);                 // ds_load_b128
      v16h bb = ldh16(&vt[n0 + nlane][k0 + kcol]);             // ds_load_b128
      acc = wmma32(a, bb, acc);
    }
  }
  const int rb = (lane >> 4) << 3;
#pragma unroll
  for (int r = 0; r < 8; ++r)
    hout[(size_t)(b * L_ + r0 + rb + r) * D_ + h * HD_ + n0 + nlane] =
        (_Float16)acc[r];
}

// ---------------------------------------------------------------------------
// LayerNorm: one block per row of 768; writes f32 and/or f16 (nullable).
// ---------------------------------------------------------------------------
__global__ void k_ln(const float* __restrict__ x, const float* __restrict__ g,
                     const float* __restrict__ bt, float* __restrict__ yf,
                     _Float16* __restrict__ yh) {
  __shared__ float red[256];
  const size_t row = blockIdx.x;
  const float* xr = x + row * D_;
  float s = 0.0f;
  for (int c = threadIdx.x; c < D_; c += 256) s += xr[c];
  red[threadIdx.x] = s;
  __syncthreads();
  for (int st = 128; st > 0; st >>= 1) {
    if (threadIdx.x < st) red[threadIdx.x] += red[threadIdx.x + st];
    __syncthreads();
  }
  const float mean = red[0] / D_;
  __syncthreads();
  float v = 0.0f;
  for (int c = threadIdx.x; c < D_; c += 256) {
    float d = xr[c] - mean;
    v += d * d;
  }
  red[threadIdx.x] = v;
  __syncthreads();
  for (int st = 128; st > 0; st >>= 1) {
    if (threadIdx.x < st) red[threadIdx.x] += red[threadIdx.x + st];
    __syncthreads();
  }
  const float inv = rsqrtf(red[0] / D_ + 1e-5f);
  for (int c = threadIdx.x; c < D_; c += 256) {
    float o = (xr[c] - mean) * inv * g[c] + bt[c];
    if (yf) yf[row * D_ + c] = o;
    if (yh) yh[row * D_ + c] = (_Float16)o;
  }
}

// tsum[b,l,:] = A[b,off+l,:] + B[b,off+l,:]; writes f32 (gate) + f16 (GEMM)
__global__ void k_addtok(const float* __restrict__ A, const float* __restrict__ Bm,
                         float* __restrict__ Df, _Float16* __restrict__ Dh,
                         int Lsrc, int off, int Ltok) {
  size_t idx = (size_t)blockIdx.x * 256 + threadIdx.x;
  size_t total = (size_t)B_ * Ltok * D_;
  if (idx >= total) return;
  int c = (int)(idx % D_);
  size_t t = idx / D_;
  int b = (int)(t / Ltok), l = (int)(t % Ltok);
  size_t sidx = ((size_t)(b * Lsrc + off + l)) * D_ + c;
  float v = A[sidx] + Bm[sidx];
  Df[idx] = v;
  Dh[idx] = (_Float16)v;
}

// gate logits + top2 softmax weights + (optional) importance/load accum
__global__ void k_gate(const float* __restrict__ t, const float* __restrict__ gw,
                       float* __restrict__ logits_out, int* __restrict__ tidx,
                       float* __restrict__ tw, float* __restrict__ accum) {
  __shared__ float red[E_][64];
  const int tok = blockIdx.x;
  const float* tr = t + (size_t)tok * D_;
  float part[E_] = {};
  for (int c = threadIdx.x; c < D_; c += 64) {
    float xv = tr[c];
#pragma unroll
    for (int e = 0; e < E_; ++e) part[e] += xv * gw[c * E_ + e];
  }
  for (int e = 0; e < E_; ++e) red[e][threadIdx.x] = part[e];
  __syncthreads();
  if (threadIdx.x == 0) {
    float lg[E_];
    for (int e = 0; e < E_; ++e) {
      float s = 0.0f;
      for (int i = 0; i < 64; ++i) s += red[e][i];
      lg[e] = s;
    }
    if (logits_out)
      for (int e = 0; e < E_; ++e) logits_out[(size_t)tok * E_ + e] = lg[e];
    int i0 = 0;
    for (int e = 1; e < E_; ++e) if (lg[e] > lg[i0]) i0 = e;
    int i1 = (i0 == 0) ? 1 : 0;
    for (int e = 0; e < E_; ++e) if (e != i0 && lg[e] > lg[i1]) i1 = e;
    float e1 = __expf(lg[i1] - lg[i0]);
    float sw = 1.0f + e1;
    tidx[2 * tok] = i0; tidx[2 * tok + 1] = i1;
    tw[2 * tok] = 1.0f / sw; tw[2 * tok + 1] = e1 / sw;
    if (accum) {
      float mm = lg[0];
      for (int e = 1; e < E_; ++e) mm = fmaxf(mm, lg[e]);
      float pe[E_], se = 0.0f;
      for (int e = 0; e < E_; ++e) { pe[e] = __expf(lg[e] - mm); se += pe[e]; }
      for (int e = 0; e < E_; ++e) atomicAdd(&accum[e], pe[e] / se);
      atomicAdd(&accum[E_ + i0], 1.0f);
      atomicAdd(&accum[E_ + i1], 1.0f);
    }
  }
}

// expert_w [E, D, MH] f32 -> packed TRANSPOSED f16 [E*MH][D]
__global__ void k_pack(const float* __restrict__ ew, _Float16* __restrict__ pk,
                       int n) {
  int i = blockIdx.x * 256 + threadIdx.x;          // i over (E*MH)*D
  if (i >= n) return;
  int nrow = i / D_, d = i % D_;
  int e = nrow / MH_, j = nrow % MH_;
  pk[i] = (_Float16)ew[((size_t)e * D_ + d) * MH_ + j];
}

// f32 [K][N] -> f16 transposed [N][K]
__global__ void k_f2h_t(const float* __restrict__ in, _Float16* __restrict__ out,
                        int K, int N) {
  int i = blockIdx.x * 256 + threadIdx.x;
  if (i >= K * N) return;
  int k = i / N, n = i % N;
  out[(size_t)n * K + k] = (_Float16)in[i];
}

// out[t,:] = w0*h[t,e0*64:+64] + w1*h[t,e1*64:+64]  (f16 out -> ffn GEMM)
__global__ void k_moecomb(const float* __restrict__ h, const int* __restrict__ tidx,
                          const float* __restrict__ tw,
                          _Float16* __restrict__ out, int ntok) {
  int idx = blockIdx.x * 256 + threadIdx.x;
  if (idx >= ntok * MH_) return;
  int t = idx / MH_, j = idx % MH_;
  size_t hb = (size_t)t * (E_ * MH_);
  out[idx] = (_Float16)(tw[2 * t] * h[hb + tidx[2 * t] * MH_ + j] +
                        tw[2 * t + 1] * h[hb + tidx[2 * t + 1] * MH_ + j]);
}

// candidate elimination: masked-mean attn -> stable descending rank sort
__global__ void k_ce(const float* __restrict__ attn,
                     const unsigned char* __restrict__ mask,
                     const int* __restrict__ gidx, int* __restrict__ order,
                     float* __restrict__ keep_out, float* __restrict__ rem_out) {
  __shared__ float at[LS_];
  __shared__ int ord[LS_];
  const int b = blockIdx.x, j = threadIdx.x;
  float msum = 0.0f;
  for (int i = 0; i < LT_; ++i) msum += mask[b * LT_ + i] ? 1.0f : 0.0f;
  float s = 0.0f;
  for (int h = 0; h < H_; ++h) {
    size_t base = (size_t)(b * H_ + h) * L_ * L_;
    float hs = 0.0f;
    for (int i = 0; i < LT_; ++i)
      if (mask[b * LT_ + i]) hs += attn[base + (size_t)i * L_ + LT_ + j];
    s += hs / msum;
  }
  at[j] = s / H_;
  __syncthreads();
  const float v = at[j];
  int rank = 0;
  for (int q = 0; q < LS_; ++q) {
    float u = at[q];
    rank += (u > v) || (u == v && q < j);
  }
  ord[rank] = j;
  __syncthreads();
  order[b * LS_ + j] = ord[j];
  float gv = (float)gidx[b * LS_ + ord[j]];
  if (j < KEEP_) keep_out[b * KEEP_ + j] = gv;
  else           rem_out[b * (LS_ - KEEP_) + (j - KEEP_)] = gv;
}

// ---------------------------------------------------------------------------
// CE token gather as an async LDS round-trip (gfx1250 ASYNCcnt path).
// ---------------------------------------------------------------------------
__global__ void k_gather(const float* __restrict__ src, const int* __restrict__ order,
                         float* __restrict__ dst) {
  __shared__ float buf[D_];
  const int gl = blockIdx.x;
  const int b = gl / L2_, l = gl % L2_;
  const int srow =
      (l < LT_) ? (b * L_ + l) : (b * L_ + LT_ + order[b * LS_ + (l - LT_)]);
  const int t = threadIdx.x;
  const unsigned ldsoff = (unsigned)(size_t)&buf[t * 4];
  const unsigned long long ga =
      (unsigned long long)(src + (size_t)srow * D_ + t * 4);
  asm volatile("global_load_async_to_lds_b128 %0, %1, off"
               :: "v"(ldsoff), "v"(ga) : "memory");
  asm volatile("s_wait_asynccnt 0" ::: "memory");
  __syncthreads();
  const unsigned long long gd =
      (unsigned long long)(dst + (size_t)gl * D_ + t * 4);
  asm volatile("global_store_async_from_lds_b128 %0, %1, off"
               :: "v"(gd), "v"(ldsoff) : "memory");
}

// out = a + b + concat(z-part, x-part)  (swap=1 -> x-part first)
__global__ void k_ccadd(const float* __restrict__ a, const float* __restrict__ b2,
                        const float* __restrict__ zp, const float* __restrict__ xp,
                        float* __restrict__ out, int Lrow, int Lz, int swap) {
  size_t idx = (size_t)blockIdx.x * 256 + threadIdx.x;
  size_t total = (size_t)B_ * Lrow * D_;
  if (idx >= total) return;
  int c = (int)(idx % D_);
  size_t t = idx / D_;
  int bb = (int)(t / Lrow), l = (int)(t % Lrow);
  int Lx = Lrow - Lz;
  float extra;
  if (!swap)
    extra = (l < Lz) ? zp[((size_t)(bb * Lz + l)) * D_ + c]
                     : xp[((size_t)(bb * Lx + (l - Lz))) * D_ + c];
  else
    extra = (l < Lx) ? xp[((size_t)(bb * Lx + l)) * D_ + c]
                     : zp[((size_t)(bb * Lz + (l - Lx))) * D_ + c];
  out[idx] = a[idx] + b2[idx] + extra;
}

__global__ void k_i2f(const int* __restrict__ in, float* __restrict__ out, int n) {
  int i = blockIdx.x * 256 + threadIdx.x;
  if (i < n) out[i] = (float)in[i];
}

// loss = sum over 4 moe calls of E * sum_e importance_e * load_e
__global__ void k_loss(const float* __restrict__ acc, float* __restrict__ out) {
  const float n[4] = {(float)(B_ * LS_), (float)(B_ * LT_),
                      (float)(B_ * (L2_ - LT2_)), (float)(B_ * LT2_)};
  float loss = 0.0f;
  for (int s4 = 0; s4 < 4; ++s4) {
    const float* a = acc + s4 * 12;
    for (int e = 0; e < E_; ++e)
      loss += (float)E_ * (a[e] / n[s4]) * (a[E_ + e] / n[s4]);
  }
  *out = loss;
}

// ---------------------------------------------------------------------------
// Host-side MoE "prompt half": tsum -> gate -> expert GEMM -> combine -> ffn1
// ---------------------------------------------------------------------------
static void run_moe(const float* a, const float* b, int Lsrc, int off, int Ltok,
                    float* tsumF, _Float16* tsumH, float* hall, _Float16* moeoH,
                    int* tidx, float* tw, const float* gate_w,
                    const _Float16* packE, const _Float16* ffnWT,
                    const float* ffn1_b, float* outbuf, float* logits_out,
                    float* accum, hipStream_t s) {
  const int ntok = B_ * Ltok;                            // multiple of 16
  const size_t tot = (size_t)ntok * D_;
  k_addtok<<<(unsigned)((tot + 255) / 256), 256, 0, s>>>(a, b, tsumF, tsumH,
                                                         Lsrc, off, Ltok);
  k_gate<<<ntok, 64, 0, s>>>(tsumF, gate_w, logits_out, tidx, tw, accum);
  k_gemm<0, 0><<<dim3((E_ * MH_ + 255) / 256, ntok / 16), 128, 0, s>>>(
      tsumH, packE, nullptr, hall, nullptr, ntok, E_ * MH_, D_);
  const int nc = ntok * MH_;
  k_moecomb<<<(nc + 255) / 256, 256, 0, s>>>(hall, tidx, tw, moeoH, ntok);
  k_gemm<0, 0><<<dim3((D_ + 255) / 256, ntok / 16), 128, 0, s>>>(
      moeoH, ffnWT, ffn1_b, outbuf, nullptr, ntok, D_, MH_);
}

extern "C" void kernel_launch(void* const* d_in, const int* in_sizes, int n_in,
                              void* d_out_v, int out_size, void* d_ws,
                              size_t ws_size, hipStream_t stream) {
  (void)in_sizes; (void)n_in; (void)out_size; (void)ws_size;
  const float* x    = (const float*)d_in[0];
  const float* xi   = (const float*)d_in[1];
  const int* git    = (const int*)d_in[2];
  const int* giti   = (const int*)d_in[3];
  const int* gis    = (const int*)d_in[4];
  const int* gisi   = (const int*)d_in[5];
  const unsigned char* maskz = (const unsigned char*)d_in[6];
  const float* n1_g  = (const float*)d_in[7];
  const float* n1_b  = (const float*)d_in[8];
  const float* n1m_g = (const float*)d_in[9];
  const float* n1m_b = (const float*)d_in[10];
  const float* n2_g  = (const float*)d_in[11];
  const float* n2_b  = (const float*)d_in[12];
  const float* n2m_g = (const float*)d_in[13];
  const float* n2m_b = (const float*)d_in[14];
  const float* qkv_w = (const float*)d_in[15];
  const float* proj_w = (const float*)d_in[16];
  const float* proj_b = (const float*)d_in[17];
  const float* fc1_w = (const float*)d_in[18];
  const float* fc1_b = (const float*)d_in[19];
  const float* fc2_w = (const float*)d_in[20];
  const float* fc2_b = (const float*)d_in[21];
  const float* gate_w = (const float*)d_in[22];
  const float* expert_w = (const float*)d_in[23];
  const float* ffn1_w = (const float*)d_in[24];
  const float* ffn1_b = (const float*)d_in[25];
  float* outp = (float*)d_out_v;

  // output offsets (floats) in reference return order
  const size_t XOUT = (size_t)B_ * L2_ * D_;
  const size_t ATT  = (size_t)B_ * H_ * L_ * L_;
  const size_t o_x    = 0;
  const size_t o_git  = o_x + XOUT;
  const size_t o_gis  = o_git + (size_t)B_ * LT_;
  const size_t o_rem  = o_gis + (size_t)B_ * KEEP_;
  const size_t o_attn = o_rem + (size_t)B_ * (LS_ - KEEP_);
  const size_t o_xi   = o_attn + ATT;
  const size_t o_giti = o_xi + XOUT;
  const size_t o_gisi = o_giti + (size_t)B_ * LT_;
  const size_t o_remi = o_gisi + (size_t)B_ * KEEP_;
  const size_t o_iatt = o_remi + (size_t)B_ * (LS_ - KEEP_);
  const size_t o_loss = o_iatt + ATT;
  const size_t o_la   = o_loss + 1;
  const size_t o_lf   = o_la + (size_t)B_ * LS_ * E_;

  // workspace carve-out (float-granular)
  float* w = (float*)d_ws;
  size_t off = 0;
  auto alloc  = [&](size_t n) { float* p = w + off; off += n; return p; };
  auto allocH = [&](size_t n) {
    _Float16* p = (_Float16*)(w + off); off += (n + 1) / 2; return p;
  };
  const size_t TOK1 = (size_t)B_ * L_;   // 10240
  const size_t TOK2 = (size_t)B_ * L2_;  // 7808
  const size_t BIGH = (TOK2 * FH_ > TOK1 * QW_) ? TOK2 * FH_ : TOK1 * QW_;
  float* xn1  = alloc(TOK1 * D_);
  float* xn1i = alloc(TOK1 * D_);
  float* xat  = alloc(TOK1 * D_);
  float* x1   = alloc(TOK1 * D_);
  float* xi1  = alloc(TOK1 * D_);
  float* tsumF = alloc((size_t)B_ * LS_ * D_);
  float* hall  = alloc((size_t)B_ * LS_ * (E_ * MH_));
  float* pXP   = alloc((size_t)B_ * LS_ * D_);
  float* pZP   = alloc((size_t)B_ * LT_ * D_);
  float* xce  = alloc(TOK2 * D_);
  float* xice = alloc(TOK2 * D_);
  float* xn2  = alloc(TOK2 * D_);
  float* xn2i = alloc(TOK2 * D_);
  float* tw     = alloc((size_t)B_ * LS_ * 2);
  float* accums = alloc(48);
  int* tidx  = (int*)alloc((size_t)B_ * LS_ * 2);
  int* ordX  = (int*)alloc((size_t)B_ * LS_);
  int* ordXi = (int*)alloc((size_t)B_ * LS_);
  _Float16* lnAh  = allocH(TOK1 * D_);
  _Float16* bigh  = allocH(BIGH);               // qkv out / fc1 out
  _Float16* aouth = allocH(TOK1 * D_);
  _Float16* tsumH = allocH((size_t)B_ * LS_ * D_);
  _Float16* moeoH = allocH((size_t)B_ * LS_ * MH_);
  _Float16* wqkvT = allocH((size_t)D_ * QW_);   // [2304][768]
  _Float16* wprojT = allocH((size_t)D_ * D_);   // [768][768]
  _Float16* wfc1T = allocH((size_t)D_ * FH_);   // [3072][768]
  _Float16* wfc2T = allocH((size_t)FH_ * D_);   // [768][3072]
  _Float16* wffnT = allocH((size_t)MH_ * D_);   // [768][64]
  _Float16* packE = allocH((size_t)D_ * E_ * MH_);  // [384][768]

  hipMemsetAsync(accums, 0, 48 * sizeof(float), stream);
  // one-time f32 -> transposed-f16 weight conversions
  {
    auto cvtT = [&](const float* src, _Float16* dst, int K, int N) {
      int n = K * N;
      k_f2h_t<<<(n + 255) / 256, 256, 0, stream>>>(src, dst, K, N);
    };
    cvtT(qkv_w, wqkvT, D_, QW_);
    cvtT(proj_w, wprojT, D_, D_);
    cvtT(fc1_w, wfc1T, D_, FH_);
    cvtT(fc2_w, wfc2T, FH_, D_);
    cvtT(ffn1_w, wffnT, MH_, D_);
    int n = E_ * MH_ * D_;
    k_pack<<<(n + 255) / 256, 256, 0, stream>>>(expert_w, packE, n);
  }

  // ------------------------- phase 1 -------------------------
  k_ln<<<(unsigned)TOK1, 256, 0, stream>>>(x,  n1m_g, n1m_b, xn1,  nullptr);
  k_ln<<<(unsigned)TOK1, 256, 0, stream>>>(xi, n1m_g, n1m_b, xn1i, nullptr);
  for (int br = 0; br < 2; ++br) {
    const float* resid = br ? xi : x;
    k_ln<<<(unsigned)TOK1, 256, 0, stream>>>(resid, n1_g, n1_b, nullptr, lnAh);
    k_gemm<0, 1><<<dim3((QW_ + 255) / 256, TOK1 / 16), 128, 0, stream>>>(
        lnAh, wqkvT, nullptr, nullptr, bigh, (int)TOK1, QW_, D_);
    k_attn<<<dim3(L_ / 16, H_, B_), 128, 0, stream>>>(
        bigh, aouth, outp + (br ? o_iatt : o_attn));
    k_gemm<0, 0><<<dim3((D_ + 255) / 256, TOK1 / 16), 128, 0, stream>>>(
        aouth, wprojT, proj_b, xat, nullptr, (int)TOK1, D_, D_);
    const float* pa = br ? xn1i : xn1;
    const float* pb = br ? xn1 : xn1i;
    run_moe(pa, pb, L_, LT_, LS_, tsumF, tsumH, hall, moeoH, tidx, tw, gate_w,
            packE, wffnT, ffn1_b, pXP, br ? nullptr : outp + o_la,
            br ? nullptr : accums + 0, stream);
    run_moe(pa, pb, L_, 0, LT_, tsumF, tsumH, hall, moeoH, tidx, tw, gate_w,
            packE, wffnT, ffn1_b, pZP, nullptr, br ? nullptr : accums + 12,
            stream);
    size_t tot = TOK1 * D_;
    k_ccadd<<<(unsigned)((tot + 255) / 256), 256, 0, stream>>>(
        resid, xat, pZP, pXP, br ? xi1 : x1, L_, LT_, 0);
  }

  // ------------------------- CE -------------------------
  k_ce<<<B_, LS_, 0, stream>>>(outp + o_attn, maskz, gis, ordX,
                               outp + o_gis, outp + o_rem);
  k_ce<<<B_, LS_, 0, stream>>>(outp + o_iatt, maskz, gisi, ordXi,
                               outp + o_gisi, outp + o_remi);
  k_gather<<<B_ * L2_, 192, 0, stream>>>(x1, ordX, xce);
  k_gather<<<B_ * L2_, 192, 0, stream>>>(xi1, ordXi, xice);
  k_i2f<<<(B_ * LT_ + 255) / 256, 256, 0, stream>>>(git, outp + o_git, B_ * LT_);
  k_i2f<<<(B_ * LT_ + 255) / 256, 256, 0, stream>>>(giti, outp + o_giti, B_ * LT_);

  // ------------------------- phase 2 -------------------------
  k_ln<<<(unsigned)TOK2, 256, 0, stream>>>(xce,  n2m_g, n2m_b, xn2,  nullptr);
  k_ln<<<(unsigned)TOK2, 256, 0, stream>>>(xice, n2m_g, n2m_b, xn2i, nullptr);
  for (int br = 0; br < 2; ++br) {
    const float* src = br ? xice : xce;
    k_ln<<<(unsigned)TOK2, 256, 0, stream>>>(src, n2_g, n2_b, nullptr, lnAh);
    k_gemm<1, 1><<<dim3((FH_ + 255) / 256, TOK2 / 16), 128, 0, stream>>>(
        lnAh, wfc1T, fc1_b, nullptr, bigh, (int)TOK2, FH_, D_);
    k_gemm<0, 0><<<dim3((D_ + 255) / 256, TOK2 / 16), 128, 0, stream>>>(
        bigh, wfc2T, fc2_b, xat, nullptr, (int)TOK2, D_, FH_);
    const float* pa = br ? xn2i : xn2;
    const float* pb = br ? xn2 : xn2i;
    run_moe(pa, pb, L2_, LT2_, L2_ - LT2_, tsumF, tsumH, hall, moeoH, tidx, tw,
            gate_w, packE, wffnT, ffn1_b, pXP, br ? nullptr : outp + o_lf,
            br ? nullptr : accums + 24, stream);
    run_moe(pa, pb, L2_, 0, LT2_, tsumF, tsumH, hall, moeoH, tidx, tw, gate_w,
            packE, wffnT, ffn1_b, pZP, nullptr, br ? nullptr : accums + 36,
            stream);
    size_t tot = TOK2 * D_;
    k_ccadd<<<(unsigned)((tot + 255) / 256), 256, 0, stream>>>(
        src, xat, pZP, pXP, outp + (br ? o_xi : o_x), L2_, LT2_, br);
  }
  k_loss<<<1, 1, 0, stream>>>(accums, outp + o_loss);
}
